// Model_47476568490424
// MI455X (gfx1250) — matrix operand order repaired and checked by reading, not compiled, not run on hardware
//
#include <hip/hip_runtime.h>
#include <hip/hip_bf16.h>

typedef __attribute__((ext_vector_type(16))) _Float16 v16h;
typedef __attribute__((ext_vector_type(8)))  float    v8f;
typedef int v2i __attribute__((ext_vector_type(2)));

#define AS1 __attribute__((address_space(1)))
#define AS3 __attribute__((address_space(3)))

#if defined(__has_builtin)
#if __has_builtin(__builtin_amdgcn_global_load_async_to_lds_b64) && \
    __has_builtin(__builtin_amdgcn_s_wait_asynccnt)
#define HAVE_ASYNC 1
#endif
#if __has_builtin(__builtin_amdgcn_tensor_load_to_lds) && \
    __has_builtin(__builtin_amdgcn_s_wait_tensorcnt)
#define HAVE_TDM 1
#endif
#endif
#ifndef HAVE_ASYNC
#define HAVE_ASYNC 0
#endif
#ifndef HAVE_TDM
#define HAVE_TDM 0
#endif

#define DEV __device__ __forceinline__

DEV float leakyf(float v) { return v >= 0.f ? v : 0.01f * v; }

// single-instruction lane XOR via ds_swizzle_b32 (group-of-32: and=0x1f, or=0, xor=M)
template<int M>
DEV float swz_xor(float v) {
  return __int_as_float(
      __builtin_amdgcn_ds_swizzle(__float_as_int(v), (0x1f << 10) | M));
}

// cooperative global_prefetch_b8 of a byte range
DEV void prefetch_bytes(const void* p, int bytes, int tid, int nthreads) {
  const char* c = (const char*)p;
  for (int off = tid * 128; off < bytes; off += nthreads * 128)
    __builtin_prefetch(c + off, 0, 1);
}

#if HAVE_ASYNC
// async DMA: 8 bytes global -> LDS, tracked by ASYNCcnt
DEV void async_g2l_b64(void* lds_dst, const void* gsrc) {
  __builtin_amdgcn_global_load_async_to_lds_b64(
      (AS1 v2i*)(uintptr_t)gsrc,
      (AS3 v2i*)(unsigned)(uintptr_t)lds_dst,
      0, 0);
}
#endif

// ---------------- WMMA fragment loaders (gfx1250 wave32 layouts) ----------------
// A fragment: 16x32 f16 tile from row-major [rows][ldk].
DEV v16h load_afrag(const _Float16* __restrict__ A, int ldk, int row0, int k0, int lane) {
  const int m  = row0 + (lane & 15);
  const int kb = (lane >> 4) << 4;
  const _Float16* p = A + (size_t)m * ldk + k0 + kb;
  v16h a;
#pragma unroll
  for (int e = 0; e < 8; ++e) a[e] = p[e];
#pragma unroll
  for (int e = 0; e < 8; ++e) a[8 + e] = p[8 + e];
  return a;
}

// B fragment: 32x16 tile where B[k][n] = W[n][k], W row-major [ncols][ldk].
DEV v16h load_bfrag_T(const _Float16* __restrict__ W, int ldk, int n0, int k0, int lane) {
  const int n  = n0 + (lane & 15);
  const int kb = (lane >> 4) << 4;
  const _Float16* p = W + (size_t)n * ldk + k0 + kb;
  v16h b;
#pragma unroll
  for (int e = 0; e < 16; ++e) b[e] = p[e];
  return b;
}

// B fragment for im2col conv with c-major K-order: k = c*TPAD + s.
// Consecutive e -> consecutive s -> contiguous LDS (vectorizes to ds_load_b128).
template<int TPAD, int OFF>
DEV v16h load_bfrag_conv(const _Float16* xs, int ldx, int q0, int k0, int lane) {
  const int n  = lane & 15;
  const int kb = (lane >> 4) << 4;
  v16h b;
#pragma unroll
  for (int e = 0; e < 16; ++e) {
    const int kg = k0 + kb + e;
    const int c  = kg / TPAD;
    const int s  = kg % TPAD;
    b[e] = xs[c * ldx + q0 + n + s + OFF];
  }
  return b;
}

DEV v8f wmma_f16(v16h a, v16h b, v8f c) {
  return __builtin_amdgcn_wmma_f32_16x16x32_f16(false, a, false, b, (short)0, c, false, false);
}

// ---------------- fused conv (+bias, activation, pool) via WMMA ----------------
// in: f16 [B][CIN][Lin]; wA: f16 [COUTPAD][CIN*TPAD] with k = c*TPAD + s;
// out: f16 [B][COUT][Lout].
template<int CIN, int TAPS, int TPAD, int COUT, int COUTPAD, int NT,
         int POOL, int ACT, bool MAXPOOL, bool REFLECT>
__global__ void __launch_bounds__(256) conv_wmma_kernel(
    const _Float16* __restrict__ in, const _Float16* __restrict__ wA,
    const float* __restrict__ bias, _Float16* __restrict__ out,
    int Lin, int Lout)
{
  constexpr int WAVES = 8;
  constexpr int POSB  = WAVES * NT * 16;
  constexpr int K     = CIN * TPAD;
  constexpr int KST   = K / 32;
  constexpr int MT    = COUTPAD / 16;
  constexpr int P     = REFLECT ? (TAPS - 1) / 2 : 0;
  constexpr int SH    = REFLECT ? 8 : 0;            // aligned left margin
  constexpr int OFF   = SH - P;                     // xs index = q + s + OFF
  constexpr int STAGEW = POSB + (REFLECT ? 24 : 8); // covers [t0-SH, ...)
  constexpr int LDX   = STAGEW;                     // multiple of 4 -> 8B rows
  static_assert(K % 32 == 0, "K multiple of 32");

  __shared__ _Float16 xs[CIN * LDX];

  const int b    = blockIdx.y;
  const int t0   = blockIdx.x * POSB;
  const int tid  = threadIdx.x;
  const int lane = tid & 31;
  const int wave = tid >> 5;

  prefetch_bytes(wA, COUTPAD * K * 2, tid, 256);

  // stage input tile: async DMA for aligned in-bounds 4-element chunks,
  // element-wise reflect/clamp fixup at the edges.
  const _Float16* xin = in + (size_t)b * CIN * Lin;
  const int talign0 = t0 - SH;
  constexpr int CH_PER_ROW = STAGEW / 4;
  for (int i = tid; i < CIN * CH_PER_ROW; i += 256) {
    const int c  = i / CH_PER_ROW;
    const int j4 = (i - c * CH_PER_ROW) * 4;
    const int t  = talign0 + j4;
    _Float16* ld = &xs[c * LDX + j4];
    if (t >= 0 && t + 3 < Lin) {
#if HAVE_ASYNC
      async_g2l_b64(ld, xin + (size_t)c * Lin + t);
#else
#pragma unroll
      for (int e = 0; e < 4; ++e) ld[e] = xin[(size_t)c * Lin + t + e];
#endif
    } else {
#pragma unroll
      for (int e = 0; e < 4; ++e) {
        int tt = t + e;
        if (REFLECT) {
          if (tt < 0) tt = -tt;
          if (tt >= Lin) tt = 2 * Lin - 2 - tt;
        } else {
          if (tt < 0) tt = 0;
          if (tt >= Lin) tt = Lin - 1;
        }
        ld[e] = xin[(size_t)c * Lin + tt];
      }
    }
  }
#if HAVE_ASYNC
  __builtin_amdgcn_s_wait_asynccnt(0);
#endif
  __syncthreads();

  v8f acc[NT][MT];
#pragma unroll
  for (int nt = 0; nt < NT; ++nt)
#pragma unroll
    for (int mt = 0; mt < MT; ++mt)
#pragma unroll
      for (int e = 0; e < 8; ++e) acc[nt][mt][e] = 0.f;

  for (int ks = 0; ks < KST; ++ks) {
    const int k0 = ks * 32;
    v16h bf[NT];
#pragma unroll
    for (int nt = 0; nt < NT; ++nt)
      bf[nt] = load_bfrag_conv<TPAD, OFF>(xs, LDX, (wave * NT + nt) * 16, k0, lane);
#pragma unroll
    for (int mt = 0; mt < MT; ++mt) {
      const v16h a = load_afrag(wA, K, mt * 16, k0, lane);
#pragma unroll
      for (int nt = 0; nt < NT; ++nt)
        acc[nt][mt] = wmma_f16(a, bf[nt], acc[nt][mt]);
    }
  }

  // epilogue: bias -> activation -> (optional) pool-by-4 across lanes -> store
#pragma unroll
  for (int nt = 0; nt < NT; ++nt) {
    const int gbase = t0 + (wave * NT + nt) * 16;
#pragma unroll
    for (int mt = 0; mt < MT; ++mt) {
#pragma unroll
      for (int r = 0; r < 8; ++r) {
        const int m = mt * 16 + r + ((lane >> 4) << 3);
        float v = acc[nt][mt][r];
        v += (m < COUT) ? bias[m] : 0.f;
        v = (ACT == 1) ? tanhf(v) : leakyf(v);
        if (POOL == 4) {
          const float v1 = swz_xor<1>(v);
          v = MAXPOOL ? fmaxf(v, v1) : v + v1;
          const float v2 = swz_xor<2>(v);
          v = MAXPOOL ? fmaxf(v, v2) : v + v2;
          if (!MAXPOOL) v *= 0.25f;
          if ((lane & 3) == 0 && m < COUT) {
            const int p = (gbase >> 2) + ((lane & 15) >> 2);
            if (p < Lout) out[((size_t)b * COUT + m) * Lout + p] = (_Float16)v;
          }
        } else {
          const int p = gbase + (lane & 15);
          if (m < COUT && p < Lout)
            out[((size_t)b * COUT + m) * Lout + p] = (_Float16)v;
        }
      }
    }
  }
}

// ---------------- small dense GEMM: D[M][N] = A[M][K] * Bw[N][K]^T ----------------
template<int KDIM, bool LEAKY, bool OUTF32>
__global__ void __launch_bounds__(32) gemm_wmma_kernel(
    const _Float16* __restrict__ A, const _Float16* __restrict__ Bw,
    void* __restrict__ out, int ldo)
{
  const int mt = blockIdx.x, nt = blockIdx.y;
  const int lane = threadIdx.x & 31;
  v8f acc;
#pragma unroll
  for (int e = 0; e < 8; ++e) acc[e] = 0.f;
  for (int ks = 0; ks < KDIM / 32; ++ks) {
    const v16h a = load_afrag(A, KDIM, mt * 16, ks * 32, lane);
    const v16h b = load_bfrag_T(Bw, KDIM, nt * 16, ks * 32, lane);
    acc = wmma_f16(a, b, acc);
  }
  const int n = nt * 16 + (lane & 15);
#pragma unroll
  for (int r = 0; r < 8; ++r) {
    const int m = mt * 16 + r + ((lane >> 4) << 3);
    float v = acc[r];
    if (LEAKY) v = leakyf(v);
    if (OUTF32) ((float*)out)[(size_t)m * ldo + n] = v;
    else        ((_Float16*)out)[(size_t)m * ldo + n] = (_Float16)v;
  }
}

// ---------------- top-5 (descending, ties -> lowest index) ----------------
__global__ void topk_kernel(const float* __restrict__ logits, int* __restrict__ idx)
{
  const int b = threadIdx.x;
  if (b >= 128) return;
  const float* row = logits + b * 96;
  unsigned long long u0 = 0; unsigned int u1 = 0;
  for (int k = 0; k < 5; ++k) {
    float best = -3.4e38f; int bi = 0;
    for (int j = 0; j < 96; ++j) {
      const bool used = (j < 64) ? ((u0 >> j) & 1ull) : ((u1 >> (j - 64)) & 1u);
      const float v = row[j];
      if (!used && v > best) { best = v; bi = j; }
    }
    if (bi < 64) u0 |= 1ull << bi; else u1 |= 1u << (bi - 64);
    idx[b * 5 + k] = bi;
  }
}

// ---------------- gathered grouped conv + bias + leaky + maxpool4 ----------------
// xh: f16 [128][16][12288] viewed as [128][96][2048]; idx [128][5];
// gwh: f16 [16][64][16] (tap 15 zero); zp out: f16 [128][64][508]
__global__ void __launch_bounds__(256) grouped_conv_kernel(
    const _Float16* __restrict__ xh, const int* __restrict__ idx,
    const _Float16* __restrict__ gwh, const float* __restrict__ gb0,
    _Float16* __restrict__ zp)
{
  constexpr int POSB = 256, SELW = POSB + 16, MT = 4, NT = 2, KST = 3;
  __shared__ _Float16 Asm[64 * 96];   // per-batch A: A[o][kk*16+s] = gw[g(kk)][o][s]
  __shared__ _Float16 sel[6 * SELW];  // row 5 = zero pad for K 80..95
  __shared__ int sidx[5];
  __shared__ float sn0;

  const int b = blockIdx.y;
  const int t0 = blockIdx.x * POSB;
  const int tid = threadIdx.x, lane = tid & 31, wave = tid >> 5;

  if (tid < 5) sidx[tid] = idx[b * 5 + tid];
  __syncthreads();
  if (tid == 0) {
    int c = 0;
#pragma unroll
    for (int kk = 0; kk < 5; ++kk) c += ((sidx[kk] / 6) == 0) ? 1 : 0;
    sn0 = (float)c;
  }
  for (int i = tid; i < 64 * 96; i += 256) {
    const int o = i / 96, k = i - o * 96;
    const int kk = k >> 4, s = k & 15;
    _Float16 v = (_Float16)0.f;
    if (kk < 5) v = gwh[((size_t)(sidx[kk] / 6) * 64 + o) * 16 + s];
    Asm[i] = v;
  }
  // stage the 5 selected mini-window segments (f16, 8B-aligned chunks)
  for (int i = tid; i < 6 * (SELW / 4); i += 256) {
    const int kk = i / (SELW / 4);
    const int j4 = (i - kk * (SELW / 4)) * 4;
    _Float16* ld = &sel[kk * SELW + j4];
    if (kk < 5) {
      const _Float16* g = xh + (size_t)b * 196608 + (size_t)sidx[kk] * 2048 + t0 + j4;
#if HAVE_ASYNC
      async_g2l_b64(ld, g);
#else
#pragma unroll
      for (int e = 0; e < 4; ++e) ld[e] = g[e];
#endif
    } else {
#pragma unroll
      for (int e = 0; e < 4; ++e) ld[e] = (_Float16)0.f;
    }
  }
#if HAVE_ASYNC
  __builtin_amdgcn_s_wait_asynccnt(0);
#endif
  __syncthreads();

  v8f acc[NT][MT];
#pragma unroll
  for (int nt = 0; nt < NT; ++nt)
#pragma unroll
    for (int mt = 0; mt < MT; ++mt)
#pragma unroll
      for (int e = 0; e < 8; ++e) acc[nt][mt][e] = 0.f;

  for (int ks = 0; ks < KST; ++ks) {
    const int k0 = ks * 32;
    v16h bf[NT];
#pragma unroll
    for (int nt = 0; nt < NT; ++nt) {
      const int q0 = (wave * NT + nt) * 16;
      const int n = lane & 15, kb = (lane >> 4) << 4;
#pragma unroll
      for (int e = 0; e < 16; ++e) {
        const int kg = k0 + kb + e;
        bf[nt][e] = sel[(kg >> 4) * SELW + q0 + n + (kg & 15)];
      }
    }
#pragma unroll
    for (int mt = 0; mt < MT; ++mt) {
      const v16h a = load_afrag(Asm, 96, mt * 16, k0, lane);
#pragma unroll
      for (int nt = 0; nt < NT; ++nt) acc[nt][mt] = wmma_f16(a, bf[nt], acc[nt][mt]);
    }
  }

  const float n0 = sn0;
#pragma unroll
  for (int nt = 0; nt < NT; ++nt) {
    const int gbase = t0 + (wave * NT + nt) * 16;
#pragma unroll
    for (int mt = 0; mt < MT; ++mt) {
#pragma unroll
      for (int r = 0; r < 8; ++r) {
        const int m = mt * 16 + r + ((lane >> 4) << 3);
        float v = acc[nt][mt][r] + n0 * gb0[m];
        v = leakyf(v);
        v = fmaxf(v, swz_xor<1>(v));
        v = fmaxf(v, swz_xor<2>(v));
        if ((lane & 3) == 0) {
          const int p = (gbase >> 2) + ((lane & 15) >> 2);
          if (p < 508) zp[((size_t)b * 64 + m) * 508 + p] = (_Float16)v;
        }
      }
    }
  }
}

// ---------------- ew3 conv + fused global mean (accumulated into feat) ----------
// in: f16 [128][128][125]; wA: f16 [128][1024] (k = c*8+s); feat += column sums
__global__ void __launch_bounds__(256) conv_mean_kernel(
    const _Float16* __restrict__ in, const _Float16* __restrict__ wA,
    float* __restrict__ feat)
{
  constexpr int LDX = 125, KST = 32, MT = 8, TILE = 128 * 125;
  __shared__ _Float16 xs[TILE + 32];   // exact [128][125] + zero tail
  const int b = blockIdx.x, tid = threadIdx.x, lane = tid & 31, wave = tid >> 5;
  const _Float16* xin = in + (size_t)b * TILE;

  prefetch_bytes(wA, 128 * 1024 * 2, tid, 256);

#if HAVE_TDM
  if (tid == 0) {
    // Tensor DMA descriptor (ISA 8.3/8.4): 2D tile 125x128 of 2-byte elems.
    typedef unsigned int u32x4 __attribute__((ext_vector_type(4)));
    typedef int i32x8 __attribute__((ext_vector_type(8)));
    typedef int i32x4 __attribute__((ext_vector_type(4)));
    const unsigned long long ga = (unsigned long long)(uintptr_t)xin;
    u32x4 g0;
    g0[0] = 1u;                                            // count=1, user mode
    g0[1] = (unsigned)(uintptr_t)(void*)xs;                // lds_addr (bytes)
    g0[2] = (unsigned)ga;                                  // global_addr lo
    g0[3] = ((unsigned)(ga >> 32) & 0x1FFFFFFu) | 0x80000000u; // addr hi | type=2
    i32x8 g1;
    g1[0] = 1 << 16;            // workgroup_mask=0, data_size=1 (2 bytes)
    g1[1] = 125 << 16;          // tensor_dim0 = 125 (lo16)
    g1[2] = 128 << 16;          // tensor_dim0 hi=0, tensor_dim1 = 128 (lo16)
    g1[3] = 125 << 16;          // tensor_dim1 hi=0, tile_dim0 = 125
    g1[4] = 128;                // tile_dim1 = 128, tile_dim2 = 0
    g1[5] = 125;                // tensor_dim0_stride lo32
    g1[6] = 0;                  // stride hi | dim1_stride lo
    g1[7] = 0;
    i32x4 z4; z4[0] = 0; z4[1] = 0; z4[2] = 0; z4[3] = 0;
    i32x8 z8; z8[0] = 0; z8[1] = 0; z8[2] = 0; z8[3] = 0;
    z8[4] = 0; z8[5] = 0; z8[6] = 0; z8[7] = 0;
    __builtin_amdgcn_tensor_load_to_lds(g0, g1, z4, z4, z8, 0);
    __builtin_amdgcn_s_wait_tensorcnt(0);
  }
#elif HAVE_ASYNC
  for (int i = tid; i < TILE / 4; i += 256)
    async_g2l_b64(&xs[i * 4], xin + i * 4);
  __builtin_amdgcn_s_wait_asynccnt(0);
#else
  for (int i = tid; i < TILE; i += 256) xs[i] = xin[i];
#endif
  if (tid < 32) xs[TILE + tid] = (_Float16)0.f;  // guard for padded tap reads
  __syncthreads();

  v8f acc[MT];
#pragma unroll
  for (int mt = 0; mt < MT; ++mt)
#pragma unroll
    for (int e = 0; e < 8; ++e) acc[mt][e] = 0.f;

  const int q0 = wave * 16;
  for (int ks = 0; ks < KST; ++ks) {
    const int k0 = ks * 32;
    const v16h bf = load_bfrag_conv<8, 0>(xs, LDX, q0, k0, lane);
#pragma unroll
    for (int mt = 0; mt < MT; ++mt) {
      const v16h a = load_afrag(wA, 1024, mt * 16, k0, lane);
      acc[mt] = wmma_f16(a, bf, acc[mt]);
    }
  }

  const bool valid = (q0 + (lane & 15)) < 119;
#pragma unroll
  for (int mt = 0; mt < MT; ++mt) {
#pragma unroll
    for (int r = 0; r < 8; ++r) {
      float v = valid ? acc[mt][r] : 0.f;
      v += swz_xor<1>(v);
      v += swz_xor<2>(v);
      v += swz_xor<4>(v);
      v += swz_xor<8>(v);
      if ((lane & 15) == 0) {
        const int m = mt * 16 + r + ((lane >> 4) << 3);
        atomicAdd(feat + b * 128 + m, v);
      }
    }
  }
}

__global__ void init_feat_kernel(const float* __restrict__ eb3, float* __restrict__ feat)
{
  feat[blockIdx.x * 128 + threadIdx.x] = eb3[threadIdx.x] * 119.f; // mean adds bias back
}

// ---------------- classifier: log_softmax(feat/119 @ cw^T + cb) ----------------
__global__ void __launch_bounds__(32) classifier_kernel(
    const float* __restrict__ feat, const float* __restrict__ cw,
    const float* __restrict__ cb, float* __restrict__ out)
{
  const int b = blockIdx.x, lane = threadIdx.x;
  float f[4];
#pragma unroll
  for (int i = 0; i < 4; ++i) f[i] = feat[b * 128 + lane + 32 * i] * (1.f / 119.f);
  float lg[5];
#pragma unroll
  for (int j = 0; j < 5; ++j) {
    float s = 0.f;
#pragma unroll
    for (int i = 0; i < 4; ++i) s += f[i] * cw[j * 128 + lane + 32 * i];
    s += swz_xor<1>(s);
    s += swz_xor<2>(s);
    s += swz_xor<4>(s);
    s += swz_xor<8>(s);
    s += swz_xor<16>(s);
    lg[j] = s + cb[j];
  }
  float mx = lg[0];
#pragma unroll
  for (int j = 1; j < 5; ++j) mx = fmaxf(mx, lg[j]);
  float se = 0.f;
#pragma unroll
  for (int j = 0; j < 5; ++j) se += expf(lg[j] - mx);
  const float lse = mx + logf(se);
  if (lane == 0) {
#pragma unroll
    for (int j = 0; j < 5; ++j) out[b * 5 + j] = lg[j] - lse;
  }
}

// ---------------- weight packing ----------------
// w fp32 [O][I][KS] -> f16 [OPAD][I*TPAD], k = c*TPAD + s (taps minor, contiguous)
template<int I, int KS, int TPAD>
__global__ void pack_conv_w_kernel(const float* __restrict__ w, _Float16* __restrict__ out,
                                   int O, int OPAD)
{
  const int tot = OPAD * I * TPAD;
  for (int i = blockIdx.x * 256 + threadIdx.x; i < tot; i += gridDim.x * 256) {
    const int o = i / (I * TPAD);
    const int k = i - o * (I * TPAD);
    const int c = k / TPAD;
    const int s = k - c * TPAD;
    float v = 0.f;
    if (o < O && s < KS) v = w[((size_t)o * I + c) * KS + s];
    out[i] = (_Float16)v;
  }
}

__global__ void cast_f16_kernel(const float* __restrict__ in, _Float16* __restrict__ out, int n)
{
  const int i = blockIdx.x * 256 + threadIdx.x;
  if (i < n) out[i] = (_Float16)in[i];
}

__global__ void cast_f16x4_kernel(const float* __restrict__ in, _Float16* __restrict__ out, int n4)
{
  const int i = blockIdx.x * 256 + threadIdx.x;
  if (i < n4) {
    const float4 v = ((const float4*)in)[i];
    out[i * 4 + 0] = (_Float16)v.x;
    out[i * 4 + 1] = (_Float16)v.y;
    out[i * 4 + 2] = (_Float16)v.z;
    out[i * 4 + 3] = (_Float16)v.w;
  }
}

// ---------------- launch ----------------
extern "C" void kernel_launch(void* const* d_in, const int* in_sizes, int n_in,
                              void* d_out, int out_size, void* d_ws, size_t ws_size,
                              hipStream_t stream)
{
  (void)in_sizes; (void)n_in; (void)out_size; (void)ws_size;
  const float* x   = (const float*)d_in[0];
  // d_in[1] = epoch (unused in eval forward)
  const float* hw1 = (const float*)d_in[2];
  const float* hb1 = (const float*)d_in[3];
  const float* hw2 = (const float*)d_in[4];
  const float* hb2 = (const float*)d_in[5];
  const float* hw3 = (const float*)d_in[6];
  const float* hb3 = (const float*)d_in[7];
  const float* hl1 = (const float*)d_in[8];
  const float* hl2 = (const float*)d_in[9];
  const float* gw  = (const float*)d_in[10];
  const float* gb0 = (const float*)d_in[11];
  const float* ew2 = (const float*)d_in[12];
  const float* eb2 = (const float*)d_in[13];
  const float* ew3 = (const float*)d_in[14];
  const float* eb3 = (const float*)d_in[15];
  const float* cw  = (const float*)d_in[16];
  const float* cb  = (const float*)d_in[17];
  float* out = (float*)d_out;

  char* wp = (char*)d_ws;
  auto alloc = [&](size_t bytes) { char* p = wp; wp += (bytes + 255) & ~(size_t)255; return p; };

  _Float16* xh    = (_Float16*)alloc((size_t)128 * 16 * 12288 * 2); // x as f16
  _Float16* h1h   = (_Float16*)alloc((size_t)128 * 32 * 3072 * 2);
  _Float16* h2h   = (_Float16*)alloc((size_t)128 * 64 * 768 * 2);
  _Float16* h3h   = (_Float16*)alloc((size_t)128 * 6144 * 2);
  _Float16* lat1h = (_Float16*)alloc((size_t)128 * 96 * 2);
  float*    logit = (float*)   alloc((size_t)128 * 96 * 4);
  int*      idxb  = (int*)     alloc((size_t)128 * 5 * 4);
  _Float16* zp    = (_Float16*)alloc((size_t)128 * 64 * 508 * 2);
  _Float16* e2p   = (_Float16*)alloc((size_t)128 * 128 * 125 * 2);
  float*    feat  = (float*)   alloc((size_t)128 * 128 * 4);
  _Float16* w1p   = (_Float16*)alloc((size_t)32 * 128 * 2);
  _Float16* w2p   = (_Float16*)alloc((size_t)64 * 256 * 2);
  _Float16* w3p   = (_Float16*)alloc((size_t)16 * 512 * 2);
  _Float16* gwh   = (_Float16*)alloc((size_t)16 * 64 * 16 * 2);
  _Float16* ew2p  = (_Float16*)alloc((size_t)128 * 512 * 2);
  _Float16* ew3p  = (_Float16*)alloc((size_t)128 * 1024 * 2);
  _Float16* hl1h  = (_Float16*)alloc((size_t)96 * 6144 * 2);
  _Float16* hl2h  = (_Float16*)alloc((size_t)96 * 96 * 2);

  // ---- input + weight conversion to WMMA-friendly f16 layouts ----
  cast_f16x4_kernel<<<dim3(24576), 256, 0, stream>>>(x, xh, 128 * 16 * 12288 / 4);
  pack_conv_w_kernel<16, 7, 8>  <<<dim3(16),  256, 0, stream>>>(hw1, w1p, 32, 32);
  pack_conv_w_kernel<32, 7, 8>  <<<dim3(64),  256, 0, stream>>>(hw2, w2p, 64, 64);
  pack_conv_w_kernel<64, 7, 8>  <<<dim3(32),  256, 0, stream>>>(hw3, w3p, 8, 16);
  pack_conv_w_kernel<1, 15, 16> <<<dim3(64),  256, 0, stream>>>(gw,  gwh, 1024, 1024);
  pack_conv_w_kernel<64, 7, 8>  <<<dim3(256), 256, 0, stream>>>(ew2, ew2p, 128, 128);
  pack_conv_w_kernel<128, 7, 8> <<<dim3(512), 256, 0, stream>>>(ew3, ew3p, 128, 128);
  cast_f16_kernel<<<dim3((96 * 6144 + 255) / 256), 256, 0, stream>>>(hl1, hl1h, 96 * 6144);
  cast_f16_kernel<<<dim3((96 * 96 + 255) / 256),   256, 0, stream>>>(hl2, hl2h, 96 * 96);

  // ---- hypernet: conv1 -> conv2 -> conv3(tanh) ----
  conv_wmma_kernel<16, 7, 8, 32, 32, 2, 4, 0, false, true>
      <<<dim3(48, 128), 256, 0, stream>>>(xh, w1p, hb1, h1h, 12288, 3072);
  conv_wmma_kernel<32, 7, 8, 64, 64, 2, 4, 0, false, true>
      <<<dim3(12, 128), 256, 0, stream>>>(h1h, w2p, hb2, h2h, 3072, 768);
  conv_wmma_kernel<64, 7, 8, 8, 16, 2, 1, 1, false, true>
      <<<dim3(3, 128), 256, 0, stream>>>(h2h, w3p, hb3, h3h, 768, 768);

  // ---- hypernet linears: leaky(h @ hl1^T) @ hl2^T ----
  gemm_wmma_kernel<6144, true,  false><<<dim3(8, 6), 32, 0, stream>>>(h3h,   hl1h, lat1h, 96);
  gemm_wmma_kernel<96,   false, true ><<<dim3(8, 6), 32, 0, stream>>>(lat1h, hl2h, logit, 96);

  // ---- hard top-5 + gathered grouped conv ----
  topk_kernel<<<dim3(1), 128, 0, stream>>>(logit, idxb);
  grouped_conv_kernel<<<dim3(8, 128), 256, 0, stream>>>(xh, idxb, gwh, gb0, zp);

  // ---- encoder tail ----
  conv_wmma_kernel<64, 7, 8, 128, 128, 2, 4, 0, true, false>
      <<<dim3(2, 128), 256, 0, stream>>>(zp, ew2p, eb2, e2p, 508, 125);
  init_feat_kernel<<<dim3(128), 128, 0, stream>>>(eb3, feat);
  conv_mean_kernel<<<dim3(128), 256, 0, stream>>>(e2p, ew3p, feat);
  classifier_kernel<<<dim3(128), 32, 0, stream>>>(feat, cw, cb, out);
}
